// BiMambaBlock_65146063946196
// MI455X (gfx1250) — matrix operand order repaired and checked by reading, not compiled, not run on hardware
//
#include <hip/hip_runtime.h>
#include <hip/hip_bf16.h>

// ---------------------------------------------------------------------------
// BiMamba block for MI455X (gfx1250, wave32, WMMA).
// Dense projections run through v_wmma_f32_16x16x32_bf16 (fp32 accum) with
// global->LDS staging on the ASYNC pipe (global_load_async_to_lds_b128,
// s_wait_asynccnt). Sequential SSM scan is register-resident, 128 blocks.
// ---------------------------------------------------------------------------

typedef __attribute__((ext_vector_type(16))) __bf16 v16bf;
typedef __attribute__((ext_vector_type(8)))  float  v8f;

union FragU { v16bf v; uint4 q[2]; };

__device__ __forceinline__ v8f wmma_bf16(v16bf a, v16bf b, v8f c) {
  // (neg_a, A, neg_b, B, c_mod, C, reuse_a, reuse_b)
  return __builtin_amdgcn_wmma_f32_16x16x32_bf16(false, a, false, b, (short)0, c,
                                                 false, false);
}

// 16B global -> LDS copy on the async pipe (ASYNCcnt). vdst = wave-relative
// LDS byte offset (low 32 bits of a generic shared pointer), vaddr = 64-bit
// global address.
__device__ __forceinline__ void async_ld_b128(unsigned lds_off, const void* gp) {
  asm volatile("global_load_async_to_lds_b128 %0, %1, off"
               :: "v"(lds_off), "v"(gp) : "memory");
}
__device__ __forceinline__ void wait_async0() {
  asm volatile("s_wait_asynccnt 0x0" ::: "memory");
}
__device__ __forceinline__ unsigned lds_off32(const void* p) {
  return (unsigned)(unsigned long long)p;   // shared aperture: addr[31:0] = LDS offset
}

// ---------------------------------------------------------------------------
// Tiled bf16 GEMM:  C[M,N] = A[M,K] @ Bw[N,K]^T (+ optional residual)
// Block: 256 threads (8 waves). Wave tile 32x64 (2x4 WMMA tiles, 8 wmma/step).
// Workgroup tile 64x256, BK=32, double-buffered LDS fed by async loads.
// Requires: M % 64 == 0, K % 32 == 0 (true for every stage here).
// ---------------------------------------------------------------------------
#define BM 64
#define BN 256
#define BK 32
#define LSTR 48   // bf16 elems per LDS row: 32 data + 16 pad => 96B (16B aligned)

__global__ __launch_bounds__(256) void gemm_bf16_nt(
    const __hip_bfloat16* __restrict__ A,
    const __hip_bfloat16* __restrict__ Bw,
    float* __restrict__ C,
    const float* __restrict__ residual,   // nullptr -> none
    int M, int N, int K)
{
  __shared__ __align__(16) __hip_bfloat16 sA[2][BM * LSTR];   // 12 KB
  __shared__ __align__(16) __hip_bfloat16 sB[2][BN * LSTR];   // 48 KB

  const int tid  = threadIdx.x;
  const int lane = tid & 31;
  const int wave = tid >> 5;
  const int wm   = wave & 1;   // 0..1 -> 32-row slab
  const int wn   = wave >> 1;  // 0..3 -> 64-col slab
  const int m0   = blockIdx.y * BM;
  const int n0   = blockIdx.x * BN;

  // staging assignment: 16B (8 x bf16) chunks; A: 1 chunk/thread, B: 4.
  const int lrow = tid >> 2;        // 0..63
  const int lch  = (tid & 3) * 8;   // 0,8,16,24

  auto issue = [&](int kt, int buf) {
    const int gk = kt * BK + lch;
    async_ld_b128(lds_off32(&sA[buf][lrow * LSTR + lch]),
                  A + (size_t)(m0 + lrow) * K + gk);
#pragma unroll
    for (int i = 0; i < 4; ++i) {
      const int r  = lrow + i * 64;      // 0..255
      const int gn = n0 + r;
      if (gn < N)
        async_ld_b128(lds_off32(&sB[buf][r * LSTR + lch]),
                      Bw + (size_t)gn * K + gk);
    }
  };

  // A fragment (16Mx32K bf16): lane = M row (l&15); per ISA table:
  //   e<8  -> k = (l>=16?8:0)  + e        (contiguous 16B)
  //   e>=8 -> k = 16+(l>=16?8:0)+(e-8)    (contiguous 16B)
  auto fragA = [&](int buf, int tm) -> v16bf {
    const int r  = wm * 32 + tm * 16 + (lane & 15);
    const int kb = (lane >> 4) * 8;
    const __hip_bfloat16* p = &sA[buf][r * LSTR + kb];
    FragU f;
    f.q[0] = *(const uint4*)(p);
    f.q[1] = *(const uint4*)(p + 16);
    return f.v;
  };
  // B fragment (32Kx16N bf16): lane = N col (l&15); same k order as A fragment
  auto fragB = [&](int buf, int tn) -> v16bf {
    const int r  = wn * 64 + tn * 16 + (lane & 15);
    const int kb = (lane >> 4) * 8;
    const __hip_bfloat16* p = &sB[buf][r * LSTR + kb];
    FragU f;
    f.q[0] = *(const uint4*)(p);
    f.q[1] = *(const uint4*)(p + 16);
    return f.v;
  };

  v8f acc[2][4];
#pragma unroll
  for (int i = 0; i < 2; ++i)
#pragma unroll
    for (int j = 0; j < 4; ++j) acc[i][j] = {};

  const int T = K / BK;
  issue(0, 0);
  wait_async0();
  __syncthreads();

  for (int t = 0; t < T; ++t) {
    const int cur = t & 1;
    if (t + 1 < T) issue(t + 1, cur ^ 1);
    const v16bf a0 = fragA(cur, 0);
    const v16bf a1 = fragA(cur, 1);
#pragma unroll
    for (int tn = 0; tn < 4; ++tn) {
      const v16bf b = fragB(cur, tn);
      acc[0][tn] = wmma_bf16(a0, b, acc[0][tn]);
      acc[1][tn] = wmma_bf16(a1, b, acc[1][tn]);
    }
    wait_async0();
    __syncthreads();
  }

  // C/D layout: vgpr r -> m = r + (lane>=16 ? 8 : 0); n = lane&15
  const int mb0 = m0 + wm * 32 + ((lane >> 4) << 3);
  const int nb0 = n0 + wn * 64 + (lane & 15);
  if (residual != nullptr) {
#pragma unroll
    for (int tm = 0; tm < 2; ++tm)
#pragma unroll
      for (int tn = 0; tn < 4; ++tn) {
        const int nn = nb0 + tn * 16;
        if (nn < N) {
#pragma unroll
          for (int r = 0; r < 8; ++r) {
            const size_t idx = (size_t)(mb0 + tm * 16 + r) * N + nn;
            C[idx] = acc[tm][tn][r] + residual[idx];
          }
        }
      }
  } else {
#pragma unroll
    for (int tm = 0; tm < 2; ++tm)
#pragma unroll
      for (int tn = 0; tn < 4; ++tn) {
        const int nn = nb0 + tn * 16;
        if (nn < N) {
#pragma unroll
          for (int r = 0; r < 8; ++r) {
            const size_t idx = (size_t)(mb0 + tm * 16 + r) * N + nn;
            C[idx] = acc[tm][tn][r];
          }
        }
      }
  }
}

// ---------------------------------------------------------------------------
// RMSNorm of x (rows of 512) -> bf16 activations
// ---------------------------------------------------------------------------
__global__ __launch_bounds__(128) void rmsnorm_in_kernel(
    const float* __restrict__ x, const float* __restrict__ w,
    __hip_bfloat16* __restrict__ hbf, int D)
{
  const int row = blockIdx.x;
  const int tid = threadIdx.x;
  const float4 v = ((const float4*)(x + (size_t)row * D))[tid];
  float ss = v.x * v.x + v.y * v.y + v.z * v.z + v.w * v.w;
  for (int o = 16; o; o >>= 1) ss += __shfl_xor(ss, o, 32);
  __shared__ float red[4];
  if ((tid & 31) == 0) red[tid >> 5] = ss;
  __syncthreads();
  const float tot = red[0] + red[1] + red[2] + red[3];
  const float sc = rsqrtf(tot / (float)D + 1e-5f);
  const float4 wv = ((const float4*)w)[tid];
  __hip_bfloat16* op = hbf + (size_t)row * D + tid * 4;
  op[0] = __float2bfloat16(v.x * sc * wv.x);
  op[1] = __float2bfloat16(v.y * sc * wv.y);
  op[2] = __float2bfloat16(v.z * sc * wv.z);
  op[3] = __float2bfloat16(v.w * sc * wv.w);
}

__global__ void f32_to_bf16_kernel(const float* __restrict__ s,
                                   __hip_bfloat16* __restrict__ d, size_t n)
{
  size_t i = (size_t)blockIdx.x * blockDim.x + threadIdx.x;
  const size_t stride = (size_t)gridDim.x * blockDim.x;
  for (; i < n; i += stride) d[i] = __float2bfloat16(s[i]);
}

// ---------------------------------------------------------------------------
// Depthwise conv (K=4, causal in direction time) + SiLU, plus dt=softplus and
// decay=exp(dt*A). rev=1 reads the sequence reversed; outputs in
// direction-local time so the scan always runs forward.
// zx row layout (2192): [ z:0..1023 | xBC:1024..2175 | dt_raw:2176..2191 ]
// ---------------------------------------------------------------------------
__global__ __launch_bounds__(256) void conv_dt_kernel(
    const float* __restrict__ zx, const float* __restrict__ cw,
    const float* __restrict__ cb, const float* __restrict__ dtb,
    const float* __restrict__ alog,
    float* __restrict__ xbc, float* __restrict__ dto, float* __restrict__ dco,
    int S, int rev)
{
  const int t = blockIdx.x, b = blockIdx.y;
  const float* basep = zx + (size_t)b * S * 2192;
  for (int c = threadIdx.x; c < 1152; c += 256) {
    float acc = cb[c];
#pragma unroll
    for (int k = 0; k < 4; ++k) {
      const int tt = t - 3 + k;
      if (tt >= 0) {
        const int s = rev ? (S - 1 - tt) : tt;
        acc = fmaf(basep[(size_t)s * 2192 + 1024 + c], cw[c * 4 + k], acc);
      }
    }
    xbc[((size_t)b * S + t) * 1152 + c] = acc / (1.f + __expf(-acc));
  }
  if (threadIdx.x < 16) {
    const int hN = threadIdx.x;
    const int s = rev ? (S - 1 - t) : t;
    const float raw = basep[(size_t)s * 2192 + 2176 + hN] + dtb[hN];
    const float dtv = (raw > 20.f) ? raw : log1pf(__expf(raw));
    dto[((size_t)b * S + t) * 16 + hN] = dtv;
    dco[((size_t)b * S + t) * 16 + hN] = __expf(-__expf(alog[hN]) * dtv);
  }
}

// ---------------------------------------------------------------------------
// SSM scan. One block per (batch, head); 256 threads hold the 64x64 state:
// thread = (p = tid>>2, n-quad = tid&3 covering 16 states). y reduced across
// the 4-lane quad with shfl_xor. Writes y + x*D in [b, t, h*64+p] layout.
// ---------------------------------------------------------------------------
__global__ __launch_bounds__(256) void scan_kernel(
    const float* __restrict__ xbc, const float* __restrict__ dt,
    const float* __restrict__ dec, const float* __restrict__ Dp,
    float* __restrict__ y, int S)
{
  const int b  = blockIdx.y;
  const int hN = blockIdx.x;
  const int tid = threadIdx.x;
  const int p  = tid >> 2;
  const int nq = tid & 3;
  const int nb = nq * 16;
  float hreg[16];
#pragma unroll
  for (int i = 0; i < 16; ++i) hreg[i] = 0.f;
  const float Dv = Dp[hN];
  for (int t = 0; t < S; ++t) {
    const size_t ro = (size_t)b * S + t;
    const float dtv = dt[ro * 16 + hN];
    const float dcv = dec[ro * 16 + hN];
    const float xv  = xbc[ro * 1152 + hN * 64 + p];
    const float dx  = dtv * xv;
    const float4* Bv = (const float4*)(xbc + ro * 1152 + 1024 + nb);
    const float4* Cv = (const float4*)(xbc + ro * 1152 + 1088 + nb);
    float ys = 0.f;
#pragma unroll
    for (int q = 0; q < 4; ++q) {
      const float4 Bq = Bv[q];
      const float4 Cq = Cv[q];
      hreg[4*q+0] = fmaf(hreg[4*q+0], dcv, dx * Bq.x); ys = fmaf(hreg[4*q+0], Cq.x, ys);
      hreg[4*q+1] = fmaf(hreg[4*q+1], dcv, dx * Bq.y); ys = fmaf(hreg[4*q+1], Cq.y, ys);
      hreg[4*q+2] = fmaf(hreg[4*q+2], dcv, dx * Bq.z); ys = fmaf(hreg[4*q+2], Cq.z, ys);
      hreg[4*q+3] = fmaf(hreg[4*q+3], dcv, dx * Bq.w); ys = fmaf(hreg[4*q+3], Cq.w, ys);
    }
    ys += __shfl_xor(ys, 1, 4);
    ys += __shfl_xor(ys, 2, 4);
    if (nq == 0) y[ro * 1024 + hN * 64 + p] = ys + xv * Dv;
  }
}

// ---------------------------------------------------------------------------
// Gated RMSNorm: out = rmsnorm(y * silu(z)) * gw, cast to bf16.
// z read from zx with direction-flipped indexing when rev=1.
// ---------------------------------------------------------------------------
__global__ __launch_bounds__(256) void gate_norm_kernel(
    const float* __restrict__ y, const float* __restrict__ zx,
    const float* __restrict__ gw, __hip_bfloat16* __restrict__ out,
    int S, int rev)
{
  const int t = blockIdx.x, b = blockIdx.y;
  const int s = rev ? (S - 1 - t) : t;
  const int tid = threadIdx.x;
  const float* yr = y  + ((size_t)b * S + t) * 1024;
  const float* zr = zx + ((size_t)b * S + s) * 2192;
  float g[4];
  float ss = 0.f;
#pragma unroll
  for (int i = 0; i < 4; ++i) {
    const int c = tid + i * 256;
    const float zv = zr[c];
    const float gv = yr[c] * (zv / (1.f + __expf(-zv)));
    g[i] = gv;
    ss += gv * gv;
  }
  for (int o = 16; o; o >>= 1) ss += __shfl_xor(ss, o, 32);
  __shared__ float red[8];
  if ((tid & 31) == 0) red[tid >> 5] = ss;
  __syncthreads();
  float tot = 0.f;
#pragma unroll
  for (int i = 0; i < 8; ++i) tot += red[i];
  const float sc = rsqrtf(tot / 1024.f + 1e-5f);
  __hip_bfloat16* op = out + ((size_t)b * S + t) * 1024;
#pragma unroll
  for (int i = 0; i < 4; ++i) {
    const int c = tid + i * 256;
    op[c] = __float2bfloat16(g[i] * sc * gw[c]);
  }
}

// concat [fo | flip(bo)] and cast to bf16 for the final projection
__global__ __launch_bounds__(512) void cat_kernel(
    const float* __restrict__ fo, const float* __restrict__ bo,
    __hip_bfloat16* __restrict__ cat, int S)
{
  const int t = blockIdx.x, b = blockIdx.y;
  const int c = threadIdx.x;  // 512 threads
  cat[((size_t)b * S + t) * 1024 + c] =
      __float2bfloat16(fo[((size_t)b * S + t) * 512 + c]);
  cat[((size_t)b * S + t) * 1024 + 512 + c] =
      __float2bfloat16(bo[((size_t)b * S + (S - 1 - t)) * 512 + c]);
}

// ---------------------------------------------------------------------------
extern "C" void kernel_launch(void* const* d_in, const int* in_sizes, int n_in,
                              void* d_out, int out_size, void* d_ws, size_t ws_size,
                              hipStream_t stream)
{
  (void)in_sizes; (void)n_in; (void)out_size; (void)ws_size;
  const int B = 4, S = 1024, BS = B * S;
  const int DM = 512, DI = 1024, NH = 16, DIP = 2192;

  const float* x      = (const float*)d_in[0];
  const float* norm_w = (const float*)d_in[1];
  const float* in_w[2]   = {(const float*)d_in[2],  (const float*)d_in[10]};
  const float* conv_w[2] = {(const float*)d_in[3],  (const float*)d_in[11]};
  const float* conv_b[2] = {(const float*)d_in[4],  (const float*)d_in[12]};
  const float* dt_b[2]   = {(const float*)d_in[5],  (const float*)d_in[13]};
  const float* A_log[2]  = {(const float*)d_in[6],  (const float*)d_in[14]};
  const float* Dp[2]     = {(const float*)d_in[7],  (const float*)d_in[15]};
  const float* gw[2]     = {(const float*)d_in[8],  (const float*)d_in[16]};
  const float* out_w[2]  = {(const float*)d_in[9],  (const float*)d_in[17]};
  const float* proj_w    = (const float*)d_in[18];

  char* base = (char*)d_ws;
  size_t off = 0;
  auto alloc = [&](size_t bytes) -> void* {
    void* p = base + off;
    off += (bytes + 255) & ~(size_t)255;
    return p;
  };

  __hip_bfloat16* hbf = (__hip_bfloat16*)alloc((size_t)BS * DM * 2);
  __hip_bfloat16* winbf[2]  = {(__hip_bfloat16*)alloc((size_t)DIP * DM * 2),
                               (__hip_bfloat16*)alloc((size_t)DIP * DM * 2)};
  __hip_bfloat16* woutbf[2] = {(__hip_bfloat16*)alloc((size_t)DM * DI * 2),
                               (__hip_bfloat16*)alloc((size_t)DM * DI * 2)};
  __hip_bfloat16* wprojbf   = (__hip_bfloat16*)alloc((size_t)DM * 1024 * 2);
  float* zxb[2]  = {(float*)alloc((size_t)BS * DIP * 4),
                    (float*)alloc((size_t)BS * DIP * 4)};
  float* xbcb[2] = {(float*)alloc((size_t)BS * 1152 * 4),
                    (float*)alloc((size_t)BS * 1152 * 4)};
  float* dtbuf[2]  = {(float*)alloc((size_t)BS * NH * 4),
                      (float*)alloc((size_t)BS * NH * 4)};
  float* decbuf[2] = {(float*)alloc((size_t)BS * NH * 4),
                      (float*)alloc((size_t)BS * NH * 4)};
  float* ybuf[2] = {(float*)alloc((size_t)BS * DI * 4),
                    (float*)alloc((size_t)BS * DI * 4)};
  __hip_bfloat16* ybf[2] = {(__hip_bfloat16*)alloc((size_t)BS * DI * 2),
                            (__hip_bfloat16*)alloc((size_t)BS * DI * 2)};
  float* ob[2] = {(float*)alloc((size_t)BS * DM * 4),
                  (float*)alloc((size_t)BS * DM * 4)};
  __hip_bfloat16* catbf = (__hip_bfloat16*)alloc((size_t)BS * 1024 * 2);

  // 1) RMSNorm(x) -> bf16 activations
  rmsnorm_in_kernel<<<dim3(BS), dim3(128), 0, stream>>>(x, norm_w, hbf, DM);

  // 2) bf16 weight casts
  f32_to_bf16_kernel<<<dim3(512), dim3(256), 0, stream>>>(in_w[0],  winbf[0],  (size_t)DIP * DM);
  f32_to_bf16_kernel<<<dim3(512), dim3(256), 0, stream>>>(in_w[1],  winbf[1],  (size_t)DIP * DM);
  f32_to_bf16_kernel<<<dim3(512), dim3(256), 0, stream>>>(out_w[0], woutbf[0], (size_t)DM * DI);
  f32_to_bf16_kernel<<<dim3(512), dim3(256), 0, stream>>>(out_w[1], woutbf[1], (size_t)DM * DI);
  f32_to_bf16_kernel<<<dim3(512), dim3(256), 0, stream>>>(proj_w,   wprojbf,   (size_t)DM * 1024);

  const dim3 blk(256);
  // 3) in-proj GEMMs: [4096,512] @ [2192,512]^T
  {
    const dim3 grid((DIP + BN - 1) / BN, BS / BM);
    gemm_bf16_nt<<<grid, blk, 0, stream>>>(hbf, winbf[0], zxb[0], nullptr, BS, DIP, DM);
    gemm_bf16_nt<<<grid, blk, 0, stream>>>(hbf, winbf[1], zxb[1], nullptr, BS, DIP, DM);
  }
  // 4) conv + dt + decay (direction-local time)
  for (int d = 0; d < 2; ++d)
    conv_dt_kernel<<<dim3(S, B), dim3(256), 0, stream>>>(
        zxb[d], conv_w[d], conv_b[d], dt_b[d], A_log[d],
        xbcb[d], dtbuf[d], decbuf[d], S, d);
  // 5) sequential scans (128 independent blocks total)
  for (int d = 0; d < 2; ++d)
    scan_kernel<<<dim3(NH, B), dim3(256), 0, stream>>>(
        xbcb[d], dtbuf[d], decbuf[d], Dp[d], ybuf[d], S);
  // 6) gated rmsnorm -> bf16
  for (int d = 0; d < 2; ++d)
    gate_norm_kernel<<<dim3(S, B), dim3(256), 0, stream>>>(
        ybuf[d], zxb[d], gw[d], ybf[d], S, d);
  // 7) out-proj GEMMs: [4096,1024] @ [512,1024]^T
  {
    const dim3 grid((DM + BN - 1) / BN, BS / BM);
    gemm_bf16_nt<<<grid, blk, 0, stream>>>(ybf[0], woutbf[0], ob[0], nullptr, BS, DM, DI);
    gemm_bf16_nt<<<grid, blk, 0, stream>>>(ybf[1], woutbf[1], ob[1], nullptr, BS, DM, DI);
  }
  // 8) concat (+unflip backward) -> bf16
  cat_kernel<<<dim3(S, B), dim3(512), 0, stream>>>(ob[0], ob[1], catbf, S);
  // 9) final projection + residual: out = x + cat @ proj_w^T
  {
    const dim3 grid((DM + BN - 1) / BN, BS / BM);
    gemm_bf16_nt<<<grid, blk, 0, stream>>>(catbf, wprojbf, (float*)d_out, x,
                                           BS, DM, 1024);
  }
}
